// EfficientGaussianRenderer_5016521802263
// MI455X (gfx1250) — compile-verified, matrix operand order checked by reading, not run on hardware
//
#include <hip/hip_runtime.h>

#define H_  256
#define W_  256
#define D_  10
#define NG  512
#define FEATN 8
#define PTOT (H_*W_*D_)        // 655360 points
#define PPAIR 24               // floats per gaussian *pair* record (96B)
#define PARAM_BYTES ((NG/2)*PPAIR*4)   // 24576 B
#define FEATB_HALFS (16*32*16)         // 16 slots * 32 lanes * 16 elems
#define FEATB_BYTES (FEATB_HALFS*2)    // 16384 B

typedef __attribute__((ext_vector_type(16))) _Float16 v16h;
typedef __attribute__((ext_vector_type(8)))  float    v8f;
typedef __attribute__((ext_vector_type(4)))  float    v4f;
typedef __attribute__((ext_vector_type(4)))  int      v4i;

#if __has_builtin(__builtin_amdgcn_global_load_async_to_lds_b128) && \
    __has_builtin(__builtin_amdgcn_s_wait_asynccnt)
#define HAVE_ASYNC_LDS 1
typedef __attribute__((address_space(1))) v4i* g_v4i_p;   // global-AS vec ptr
typedef __attribute__((address_space(3))) v4i* l_v4i_p;   // LDS-AS vec ptr
#else
#define HAVE_ASYNC_LDS 0
#endif

// ---------------------------------------------------------------------------
// Prep: per-gaussian inverse covariance (symmetric 3x3 adjugate) -> Cholesky
// factor L (inv_cov = L L^T, so mahal = ||L^T d||^2). Params stored
// pair-interleaved so render-side b128 LDS loads deliver pk-ready pairs:
//   pairRec[g/2] = [mx0,mx1, my0,my1, mz0,mz1, op0,op1,
//                   l00_0,l00_1, l10_0,l10_1, l20_0,l20_1,
//                   l11_0,l11_1, l21_0,l21_1, l22_0,l22_1, pad*4]
// featB: feature matrix pre-packed in exact WMMA B-fragment layout, with a
// ones-column at N==8 so the WMMA also emits chunk_alpha.
// ---------------------------------------------------------------------------
__global__ __launch_bounds__(256) void gs_prep_kernel(
    const float* __restrict__ means, const float* __restrict__ scales,
    const float* __restrict__ rots,  const float* __restrict__ ops,
    const float* __restrict__ feats,
    float* __restrict__ params, _Float16* __restrict__ featB)
{
  int g = blockIdx.x * blockDim.x + threadIdx.x;
  if (g >= NG) return;

  float qw = rots[g*4+0], qx = rots[g*4+1], qy = rots[g*4+2], qz = rots[g*4+3];
  float qn = rsqrtf(qw*qw + qx*qx + qy*qy + qz*qz);
  qw *= qn; qx *= qn; qy *= qn; qz *= qn;

  float sx = scales[g*3+0], sy = scales[g*3+1], sz = scales[g*3+2];

  float R00 = 1.f-2.f*(qy*qy+qz*qz), R01 = 2.f*(qx*qy-qw*qz), R02 = 2.f*(qx*qz+qw*qy);
  float R10 = 2.f*(qx*qy+qw*qz), R11 = 1.f-2.f*(qx*qx+qz*qz), R12 = 2.f*(qy*qz-qw*qx);
  float R20 = 2.f*(qx*qz-qw*qy), R21 = 2.f*(qy*qz+qw*qx), R22 = 1.f-2.f*(qx*qx+qy*qy);

  float M00=R00*sx, M01=R01*sy, M02=R02*sz;
  float M10=R10*sx, M11=R11*sy, M12=R12*sz;
  float M20=R20*sx, M21=R21*sy, M22=R22*sz;

  float a = M00*M00+M01*M01+M02*M02 + 1e-6f;
  float b = M00*M10+M01*M11+M02*M12;
  float c = M00*M20+M01*M21+M02*M22;
  float d = M10*M10+M11*M11+M12*M12 + 1e-6f;
  float e = M10*M20+M11*M21+M12*M22;
  float f = M20*M20+M21*M21+M22*M22 + 1e-6f;

  float c00 = d*f-e*e, c01 = c*e-b*f, c02 = b*e-c*d;
  float c11 = a*f-c*c, c12 = b*c-a*e, c22 = a*d-b*b;
  float id  = 1.0f / (a*c00 + b*c01 + c*c02);

  // inverse covariance (SPD, symmetric)
  float i00 = c00*id, i01 = c01*id, i02 = c02*id;
  float i11 = c11*id, i12 = c12*id, i22 = c22*id;

  // Cholesky: inv_cov = L L^T, L lower-triangular
  float l00 = sqrtf(fmaxf(i00, 1e-30f));
  float r00 = 1.0f / l00;
  float l10 = i01 * r00;
  float l20 = i02 * r00;
  float l11 = sqrtf(fmaxf(i11 - l10*l10, 1e-30f));
  float l21 = (i12 - l20*l10) / l11;
  float l22 = sqrtf(fmaxf(i22 - l20*l20 - l21*l21, 0.0f));

  float* pr = params + (g >> 1) * PPAIR + (g & 1);
  pr[0]  = means[g*3+0];  // mx
  pr[2]  = means[g*3+1];  // my
  pr[4]  = means[g*3+2];  // mz
  pr[6]  = ops[g];        // opacity
  pr[8]  = l00; pr[10] = l10; pr[12] = l20;
  pr[14] = l11; pr[16] = l21; pr[18] = l22;
  pr[20] = 0.f; pr[22] = 0.f;

  // B-fragment packing: invert the 16-bit K striping (ISA 7.12.2)
  const int slot = g >> 5;                 // chunk*2 + khalf
  const int kl   = g & 31;                 // K within the 16x16x32 WMMA
  const int s    = (kl >> 3) & 1;          // lane half (lanes 0-15 vs 16-31)
  const int el   = (kl & 7) | ((kl >> 4) << 3);  // packed-half element 0..15
  #pragma unroll
  for (int fi = 0; fi < 16; ++fi) {
    float v = (fi < FEATN) ? feats[g*FEATN + fi] : (fi == FEATN ? 1.0f : 0.0f);
    featB[((slot*32 + s*16 + fi) << 4) + el] = (_Float16)v;
  }
}

// ---------------------------------------------------------------------------
// Render: one 16-point tile per wave; per 64-gaussian chunk two independent
// V_WMMA_F32_16X16X32_F16 (separate accumulators -> no D->C hazard chain)
// contract f16 alphas against features; column N=8 is chunk_alpha.
// ---------------------------------------------------------------------------
__global__ __launch_bounds__(256) void gs_render_kernel(
    const float* __restrict__ params, const _Float16* __restrict__ featB,
    float* __restrict__ out)
{
  __shared__ float    sp[(NG/2) * PPAIR];  // 24 KB pair-interleaved params
  __shared__ _Float16 sB[FEATB_HALFS];     // 16 KB B-fragments

  const int tid = threadIdx.x;

#if HAVE_ASYNC_LDS
  // Async Global->LDS staging (bypasses VGPRs, tracked by ASYNCcnt)
  {
    char*       lp = (char*)sp;
    const char* gp = (const char*)params;
    for (int i = tid*16; i < PARAM_BYTES; i += 256*16)
      __builtin_amdgcn_global_load_async_to_lds_b128(
          (g_v4i_p)(gp + i), (l_v4i_p)(lp + i), 0, 0);
    char*       lb = (char*)sB;
    const char* gb = (const char*)featB;
    for (int i = tid*16; i < FEATB_BYTES; i += 256*16)
      __builtin_amdgcn_global_load_async_to_lds_b128(
          (g_v4i_p)(gb + i), (l_v4i_p)(lb + i), 0, 0);
    __builtin_amdgcn_s_wait_asynccnt(0);
  }
#else
  for (int i = tid; i < PARAM_BYTES/16; i += 256)
    ((v4f*)sp)[i] = ((const v4f*)params)[i];
  for (int i = tid; i < FEATB_BYTES/16; i += 256)
    ((v4f*)sB)[i] = ((const v4f*)featB)[i];
#endif
  __syncthreads();

  const int lane = tid & 31;
  const int wave = tid >> 5;
  const int hi   = lane >> 4;        // half-wave: K sub-stripe / M+8
  const int n    = lane & 15;        // A row (point) and C column (feature)
  const int tile = blockIdx.x * 128 + wave * 16;
  const int p    = tile + n;

  // coords: meshgrid(z[H], y[W], x[D], 'ij'), vector = (x, y, z)
  const int   xi = p % D_;
  const int   t  = p / D_;
  const int   yi = t % W_;
  const int   zi = t / W_;
  const float cx = -1.0f + (float)xi * (2.0f / (float)(D_ - 1));
  const float cy = -1.0f + (float)yi * (2.0f / (float)(W_ - 1));
  const float cz = -1.0f + (float)zi * (2.0f / (float)(H_ - 1));

  float acc[8], rend[8];
  #pragma unroll
  for (int r = 0; r < 8; ++r) { acc[r] = 0.0f; rend[r] = 0.0f; }

  #pragma unroll 2
  for (int chunk = 0; chunk < 8; ++chunk) {
    v8f Cacc[2];
    #pragma unroll
    for (int kh = 0; kh < 2; ++kh) {
      const int slot  = chunk*2 + kh;
      const int kbase = slot * 32;

      // B: pre-packed fragment, two ds_load_b128
      const v16h B = *(const v16h*)&sB[(slot*32 + lane) << 4];

      // A: 16 f16 alphas in WMMA layout; each v handles gaussians (g, g+1)
      v16h A;
      #pragma unroll
      for (int v = 0; v < 8; ++v) {
        const int klo = (v < 4) ? (hi * 8 + 2 * v)
                                : (16 + hi * 8 + 2 * (v - 4));   // even
        const float* pr = &sp[((kbase + klo) >> 1) * PPAIR];
        const v4f a0 = *(const v4f*)(pr);       // mx0 mx1 my0 my1
        const v4f a1 = *(const v4f*)(pr + 4);   // mz0 mz1 op0 op1
        const v4f a2 = *(const v4f*)(pr + 8);   // l00. l00. l10. l10.
        const v4f a3 = *(const v4f*)(pr + 12);  // l20. l20. l11. l11.
        const v4f a4 = *(const v4f*)(pr + 16);  // l21. l21. l22. l22.
        {  // h = 0 (even slots .x/.z)
          const float dx = cx - a0.x, dy = cy - a0.z, dz = cz - a1.x;
          float u0 = fmaf(a3.x, dz, fmaf(a2.z, dy, a2.x * dx));
          float u1 = fmaf(a4.x, dz, a3.z * dy);
          float u2 = a4.z * dz;
          float mah = fmaf(u2, u2, fmaf(u1, u1, u0 * u0));
          float av  = (mah < 9.0f) ? a1.z * __expf(-0.5f * mah) : 0.0f;
          A[2*v + 0] = (_Float16)av;
        }
        {  // h = 1 (odd slots .y/.w)
          const float dx = cx - a0.y, dy = cy - a0.w, dz = cz - a1.y;
          float u0 = fmaf(a3.y, dz, fmaf(a2.w, dy, a2.y * dx));
          float u1 = fmaf(a4.y, dz, a3.w * dy);
          float u2 = a4.w * dz;
          float mah = fmaf(u2, u2, fmaf(u1, u1, u0 * u0));
          float av  = (mah < 9.0f) ? a1.w * __expf(-0.5f * mah) : 0.0f;
          A[2*v + 1] = (_Float16)av;
        }
      }
      v8f Z = {};
      Cacc[kh] = __builtin_amdgcn_wmma_f32_16x16x32_f16(
                     false, A, false, B, (short)0, Z, false, false);
    }

    // chunk_alpha lives in column N=8: lane 8 (M=r), lane 24 (M=r+8)
    const int srcLane = hi ? 24 : 8;
    #pragma unroll
    for (int r = 0; r < 8; ++r) {
      const float s     = Cacc[0][r] + Cacc[1][r];
      const float alpha = __shfl(s, srcLane, 32);
      const float w     = (1.0f - acc[r]) * alpha;
      rend[r] += w * s;
      acc[r]  += w;
    }
  }

  // BG == 0. Lane holds feature column n for 8 rows (points).
  if (n < FEATN) {
    #pragma unroll
    for (int r = 0; r < 8; ++r) {
      const int pm = tile + r + hi * 8;
      out[pm * FEATN + n] = rend[r];
    }
  }
}

// ---------------------------------------------------------------------------
extern "C" void kernel_launch(void* const* d_in, const int* in_sizes, int n_in,
                              void* d_out, int out_size, void* d_ws, size_t ws_size,
                              hipStream_t stream) {
  (void)in_sizes; (void)n_in; (void)out_size; (void)ws_size;
  const float* means  = (const float*)d_in[0];
  const float* scales = (const float*)d_in[1];
  const float* rots   = (const float*)d_in[2];
  const float* ops    = (const float*)d_in[3];
  const float* feats  = (const float*)d_in[4];

  float*    params = (float*)d_ws;
  _Float16* featB  = (_Float16*)((char*)d_ws + PARAM_BYTES);

  gs_prep_kernel<<<(NG + 255) / 256, 256, 0, stream>>>(
      means, scales, rots, ops, feats, params, featB);
  gs_render_kernel<<<PTOT / 128, 256, 0, stream>>>(
      params, featB, (float*)d_out);
}
